// SelfAttLayer_89008902242431
// MI455X (gfx1250) — matrix operand order repaired and checked by reading, not compiled, not run on hardware
//
#include <hip/hip_runtime.h>
#include <stdint.h>

// Problem constants (B=8, N=2048, D=1024)
#define BB 8
#define NN 2048
#define DD 1024
#define MROWS (BB * NN)   // 16384 rows of X/Q/K/V
#define NC (3 * DD)       // 3072 columns of concatenated [WQ|Wk|Wv]

typedef __attribute__((ext_vector_type(16))) __bf16 v16bf;
typedef __attribute__((ext_vector_type(8)))  float  v8f;
typedef __attribute__((ext_vector_type(4)))  unsigned int u32x4;
typedef __attribute__((ext_vector_type(8)))  int i32x8;
typedef __attribute__((ext_vector_type(4)))  int i32x4;

union FragBF { v16bf v; unsigned short u[16]; uint4 q[2]; };
union AccF   { v8f v; float f[8]; };
union Pack8  { unsigned short s[8]; uint4 q; };

// ---- bf16 <-> f32 via bit manipulation (RNE), no __bf16 arithmetic -------
__device__ __forceinline__ unsigned short f2bf(float f) {
  union { float f; unsigned int u; } c; c.f = f;
  unsigned int r = c.u + 0x7FFFu + ((c.u >> 16) & 1u);
  return (unsigned short)(r >> 16);
}
__device__ __forceinline__ float bf2f(unsigned short h) {
  union { unsigned int u; float f; } c; c.u = ((unsigned int)h) << 16;
  return c.f;
}

// ---- WMMA wrapper --------------------------------------------------------
__device__ __forceinline__ v8f wmma_bf16(const FragBF& a, const FragBF& b, v8f c) {
  return __builtin_amdgcn_wmma_f32_16x16x32_bf16(false, a.v, false, b.v,
                                                 (short)0, c, false, false);
}

// ---- TDM: 2-D tile load Global -> LDS (CDNA5 ISA) --
// D# group0: [1:0]=count=1, [63:32]=lds_addr, [120:64]=global_addr, [127:126]=type=2
// D# group1: [17:16]=data_size(1=2B), [20]=pad_en, [24:22]=pad_interval,
//            [31:25]=pad_amount, [79:48]=tensor_dim0, [111:80]=tensor_dim1,
//            [127:112]=tile_dim0, [143:128]=tile_dim1, [207:160]=dim0_stride.
// Pads 4 DWORDs after every 16 DWORDs (one 32-elem row) -> LDS row stride 40
// bf16 (20 dwords), which makes the 16-lane ds_load_b128 reads conflict-free.
// This toolchain uses the 6-arg builtin form (extra int32x8, zero for 2-D).
__device__ __forceinline__ void tdm_load_tile_2d(unsigned lds_off,
                                                 const unsigned short* gptr,
                                                 unsigned tile_x, unsigned tile_y,
                                                 unsigned dim_x, unsigned dim_y,
                                                 unsigned stride_x) {
  const unsigned long long ga = (unsigned long long)(uintptr_t)gptr;
  u32x4 g0;
  g0[0] = 1u;                                          // count=1, user mode
  g0[1] = lds_off;                                     // LDS byte address
  g0[2] = (unsigned)(ga & 0xFFFFFFFFu);                // global_addr[31:0]
  g0[3] = (unsigned)((ga >> 32) & 0x01FFFFFFu) | (2u << 30);  // [56:32], type=2
  i32x8 g1;
  g1[0] = (int)((1u << 16) |                           // data_size = 2 bytes
                (1u << 20) | (3u << 22) | (3u << 25)); // pad: every 16 DW, +4 DW
  g1[1] = (int)((dim_x & 0xFFFFu) << 16);              // tensor_dim0 lo16
  g1[2] = (int)(((dim_x >> 16) & 0xFFFFu) | ((dim_y & 0xFFFFu) << 16));
  g1[3] = (int)(((dim_y >> 16) & 0xFFFFu) | ((tile_x & 0xFFFFu) << 16));
  g1[4] = (int)(tile_y & 0xFFFFu);                     // tile_dim1; tile_dim2=0
  g1[5] = (int)stride_x;                               // tensor_dim0_stride lo32
  g1[6] = 0;
  g1[7] = 0;
  i32x4 z4 = {0, 0, 0, 0};                             // 2-D: groups 2/3 unused
  i32x8 z8 = {0, 0, 0, 0, 0, 0, 0, 0};
  __builtin_amdgcn_tensor_load_to_lds(g0, g1, z4, z4, z8, 0);
}

// ---- Fragment loaders (element s holds k = 16*half + s in both A and B, contiguous order) ---------
__device__ __forceinline__ void loadA_rm1(FragBF& d, const unsigned short* __restrict__ A,
                                          size_t row0, int k0, int lda, int lane) {
  const int m = lane & 15, half = lane >> 4;
  const size_t base = (row0 + (size_t)m) * (size_t)lda + (size_t)(k0 + half * 16);
  d.q[0] = *reinterpret_cast<const uint4*>(A + base);
  d.q[1] = *reinterpret_cast<const uint4*>(A + base + 8);
}

__device__ __forceinline__ void loadB_tr1(FragBF& d, const unsigned short* __restrict__ BT,
                                          size_t base, int k0, int col0, int ldk, int lane) {
  const int n = lane & 15, half = lane >> 4;
  const size_t p = base + (size_t)(col0 + n) * (size_t)ldk + (size_t)(k0 + half * 16);
  d.q[0] = *reinterpret_cast<const uint4*>(BT + p);
  d.q[1] = *reinterpret_cast<const uint4*>(BT + p + 8);
}

__device__ __forceinline__ void storeC(float* __restrict__ C, const AccF& a,
                                       size_t row0, int col0, int ldc, int lane, float scale) {
  const int n = lane & 15, half = lane >> 4;
#pragma unroll
  for (int r = 0; r < 8; ++r)
    C[(row0 + (size_t)(half * 8 + r)) * (size_t)ldc + (size_t)(col0 + n)] = a.f[r] * scale;
}

// ===== Kernel 1a: split-convert a flat f32 array into bf16 hi/lo ==========
__global__ void cvt_split(const float* __restrict__ src,
                          unsigned short* __restrict__ hi,
                          unsigned short* __restrict__ lo, int n) {
  const int i = blockIdx.x * blockDim.x + threadIdx.x;
  if (i >= n) return;
  const float f = src[i];
  const unsigned short h = f2bf(f);
  hi[i] = h;
  lo[i] = f2bf(f - bf2f(h));
}

// ===== Kernel 1b: concat + TRANSPOSE + split [WQ|Wk|Wv] -> WT[3072,1024] ==
__global__ void cvt_w(const float* __restrict__ WQ, const float* __restrict__ Wk,
                      const float* __restrict__ Wv,
                      unsigned short* __restrict__ WhT, unsigned short* __restrict__ WlT) {
  const int i = blockIdx.x * blockDim.x + threadIdx.x;  // over NC*DD
  if (i >= DD * NC) return;
  const int c = i / DD;          // output row of WT = original column (0..3071)
  const int r = i - c * DD;      // original row (0..1023), contiguous in WT
  const float* src = (c < DD) ? WQ : ((c < 2 * DD) ? Wk : Wv);
  const float f = src[r * DD + (c & (DD - 1))];
  const unsigned short h = f2bf(f);
  WhT[i] = h;
  WlT[i] = f2bf(f - bf2f(h));
}

// ===== Kernel 2: QKV = X @ W, split-bf16, TDM-staged B panel ==============
// Block = 128(M) x 64(N), 8 waves of 32x32. Wave 0 DMAs the 64x32 B tile
// (hi+lo) into double-buffered LDS with TENSOR_LOAD_TO_LDS; everyone reads
// B fragments from LDS (ds_load_b128, conflict-free via TDM padding).
// A fragments stay direct b128 global loads (2x reuse only).
#define BROW 40   // padded LDS row stride in bf16 elements (32 data + 8 pad)
__global__ __launch_bounds__(256) void qkv_gemm(
    const unsigned short* __restrict__ Xh, const unsigned short* __restrict__ Xl,
    const unsigned short* __restrict__ WhT, const unsigned short* __restrict__ WlT,
    unsigned short* __restrict__ QhT, unsigned short* __restrict__ QlT,
    unsigned short* __restrict__ KhT, unsigned short* __restrict__ KlT,
    unsigned short* __restrict__ Vh) {
  __shared__ unsigned short Bh_s[2][64][BROW];
  __shared__ unsigned short Bl_s[2][64][BROW];

  const int lane = threadIdx.x & 31;
  const int wave = threadIdx.x >> 5;
  const int row0 = blockIdx.y * 128 + (wave >> 1) * 32;
  const int col0 = blockIdx.x * 64 + (wave & 1) * 32;
  const int colB = blockIdx.x * 64;   // block's B panel origin

  AccF acc[2][2];
#pragma unroll
  for (int i = 0; i < 2; ++i)
#pragma unroll
    for (int j = 0; j < 2; ++j)
#pragma unroll
      for (int r = 0; r < 8; ++r) acc[i][j].f[r] = 0.0f;

  // prologue: stage k-chunk 0 into buffer 0
  if (wave == 0) {
    tdm_load_tile_2d((unsigned)(uintptr_t)&Bh_s[0][0][0],
                     WhT + (size_t)colB * DD, 32, 64, DD, NC, DD);
    tdm_load_tile_2d((unsigned)(uintptr_t)&Bl_s[0][0][0],
                     WlT + (size_t)colB * DD, 32, 64, DD, NC, DD);
  }

  int buf = 0;
  for (int k0 = 0; k0 < DD; k0 += 32) {
    if (wave == 0) {
      if (k0 + 32 < DD) {
        tdm_load_tile_2d((unsigned)(uintptr_t)&Bh_s[buf ^ 1][0][0],
                         WhT + (size_t)colB * DD + (k0 + 32), 32, 64, DD, NC, DD);
        tdm_load_tile_2d((unsigned)(uintptr_t)&Bl_s[buf ^ 1][0][0],
                         WlT + (size_t)colB * DD + (k0 + 32), 32, 64, DD, NC, DD);
        __builtin_amdgcn_s_wait_tensorcnt(2);  // current buffer's 2 loads done
      } else {
        __builtin_amdgcn_s_wait_tensorcnt(0);
      }
    }
    __syncthreads();  // B tile visible to all waves

    __builtin_prefetch(Xh + (size_t)(row0 + (lane & 15)) * DD + (size_t)(k0 + 32), 0, 1);

    FragBF ah[2], al[2], bh[2], bl[2];
#pragma unroll
    for (int mt = 0; mt < 2; ++mt) {
      loadA_rm1(ah[mt], Xh, (size_t)(row0 + mt * 16), k0, DD, lane);
      loadA_rm1(al[mt], Xl, (size_t)(row0 + mt * 16), k0, DD, lane);
    }
    {
      const int n = lane & 15, half = lane >> 4;
#pragma unroll
      for (int nt = 0; nt < 2; ++nt) {
        const int r = (wave & 1) * 32 + nt * 16 + n;  // column within B panel
        const unsigned short* ph = &Bh_s[buf][r][half * 16];
        const unsigned short* pl = &Bl_s[buf][r][half * 16];
        bh[nt].q[0] = *reinterpret_cast<const uint4*>(ph);
        bh[nt].q[1] = *reinterpret_cast<const uint4*>(ph + 8);
        bl[nt].q[0] = *reinterpret_cast<const uint4*>(pl);
        bl[nt].q[1] = *reinterpret_cast<const uint4*>(pl + 8);
      }
    }
#pragma unroll
    for (int mt = 0; mt < 2; ++mt)
#pragma unroll
      for (int nt = 0; nt < 2; ++nt) {
        acc[mt][nt].v = wmma_bf16(ah[mt], bh[nt], acc[mt][nt].v);  // Ah*Bh
        acc[mt][nt].v = wmma_bf16(ah[mt], bl[nt], acc[mt][nt].v);  // Ah*Bl
        acc[mt][nt].v = wmma_bf16(al[mt], bh[nt], acc[mt][nt].v);  // Al*Bh
      }
    __syncthreads();  // all reads done before wave 0 overwrites this buffer
    buf ^= 1;
  }

  const int n = lane & 15, half = lane >> 4;
#pragma unroll
  for (int mt = 0; mt < 2; ++mt) {
    const int rowb = row0 + mt * 16 + half * 8;  // 8 consecutive rows, same batch
    const int b = rowb >> 11;                    // batch index (N=2048)
    const int n0 = rowb & (NN - 1);              // n within batch
#pragma unroll
    for (int nt = 0; nt < 2; ++nt) {
      const int col = col0 + nt * 16 + n;
      const int cc = col & (DD - 1);
      const int t = col >> 10;  // 0=Q, 1=K, 2=V
      if (t == 2) {
#pragma unroll
        for (int r = 0; r < 8; ++r)
          Vh[(size_t)(rowb + r) * DD + cc] = f2bf(acc[mt][nt].f[r]);
      } else {
        Pack8 hp, lp;
#pragma unroll
        for (int r = 0; r < 8; ++r) {
          const float cv = acc[mt][nt].f[r];
          hp.s[r] = f2bf(cv);
          lp.s[r] = f2bf(cv - bf2f(hp.s[r]));
        }
        const size_t p = ((size_t)b * DD + cc) * NN + n0;  // [b][d or e][n]
        unsigned short* dh = (t == 0) ? QhT : KhT;
        unsigned short* dl = (t == 0) ? QlT : KlT;
        *reinterpret_cast<uint4*>(dh + p) = hp.q;          // 8 contiguous bf16
        *reinterpret_cast<uint4*>(dl + p) = lp.q;
      }
    }
  }
}

// ===== Kernel 3: H[b] = K[b]^T @ Q[b] / sqrt(D), split-bf16 ===============
__global__ __launch_bounds__(256) void h_gemm(
    const unsigned short* __restrict__ KhT, const unsigned short* __restrict__ KlT,
    const unsigned short* __restrict__ QhT, const unsigned short* __restrict__ QlT,
    float* __restrict__ Hf) {
  const int lane = threadIdx.x & 31;
  const int wave = threadIdx.x >> 5;
  const int b = blockIdx.z;
  const int row0 = blockIdx.y * 128 + (wave >> 1) * 32;  // d (from K)
  const int col0 = blockIdx.x * 64 + (wave & 1) * 32;    // e (from Q)
  const size_t arow0 = (size_t)b * DD + (size_t)row0;    // absolute KT row
  const size_t qbase = (size_t)b * DD * NN;

  AccF acc[2][2];
#pragma unroll
  for (int i = 0; i < 2; ++i)
#pragma unroll
    for (int j = 0; j < 2; ++j)
#pragma unroll
      for (int r = 0; r < 8; ++r) acc[i][j].f[r] = 0.0f;

  for (int k0 = 0; k0 < NN; k0 += 32) {  // contraction over n
    __builtin_prefetch(KhT + (arow0 + (size_t)(lane & 15)) * NN + (size_t)(k0 + 32), 0, 1);

    FragBF ah[2], al[2], bh[2], bl[2];
#pragma unroll
    for (int mt = 0; mt < 2; ++mt) {
      loadA_rm1(ah[mt], KhT, arow0 + (size_t)(mt * 16), k0, NN, lane);
      loadA_rm1(al[mt], KlT, arow0 + (size_t)(mt * 16), k0, NN, lane);
    }
#pragma unroll
    for (int nt = 0; nt < 2; ++nt) {
      loadB_tr1(bh[nt], QhT, qbase, k0, col0 + nt * 16, NN, lane);
      loadB_tr1(bl[nt], QlT, qbase, k0, col0 + nt * 16, NN, lane);
    }
#pragma unroll
    for (int mt = 0; mt < 2; ++mt)
#pragma unroll
      for (int nt = 0; nt < 2; ++nt) {
        acc[mt][nt].v = wmma_bf16(ah[mt], bh[nt], acc[mt][nt].v);
        acc[mt][nt].v = wmma_bf16(ah[mt], bl[nt], acc[mt][nt].v);
        acc[mt][nt].v = wmma_bf16(al[mt], bh[nt], acc[mt][nt].v);
      }
  }

  const size_t hb = (size_t)b * DD * DD;
#pragma unroll
  for (int mt = 0; mt < 2; ++mt)
#pragma unroll
    for (int nt = 0; nt < 2; ++nt)
      storeC(Hf + hb, acc[mt][nt], (size_t)(row0 + mt * 16), col0 + nt * 16, DD,
             lane, 0.03125f /* 1/sqrt(1024) */);
}

// ===== Kernel 4: softmax over the BATCH axis; write H^T via LDS transpose =
__global__ __launch_bounds__(256) void softmax_batch(const float* __restrict__ Hf,
                                                     unsigned short* __restrict__ HbT) {
  __shared__ unsigned short tile[32][33];
  const int tx = threadIdx.x & 31;
  const int ty = threadIdx.x >> 5;
  const int d0 = (blockIdx.x >> 5) * 32;
  const int e0 = (blockIdx.x & 31) * 32;

  float p[4][BB];
#pragma unroll
  for (int rr = 0; rr < 4; ++rr) {
    const int d = d0 + ty + rr * 8;
    float v[BB];
    float m = -3.402823466e38f;
#pragma unroll
    for (int b = 0; b < BB; ++b) {
      v[b] = Hf[((size_t)b * DD + d) * DD + (size_t)(e0 + tx)];
      m = fmaxf(m, v[b]);
    }
    float s = 0.0f;
#pragma unroll
    for (int b = 0; b < BB; ++b) { v[b] = __expf(v[b] - m); s += v[b]; }
    const float inv = 1.0f / s;
#pragma unroll
    for (int b = 0; b < BB; ++b) p[rr][b] = v[b] * inv;
  }

#pragma unroll
  for (int b = 0; b < BB; ++b) {
    __syncthreads();
#pragma unroll
    for (int rr = 0; rr < 4; ++rr)
      tile[ty + rr * 8][tx] = f2bf(p[rr][b]);
    __syncthreads();
#pragma unroll
    for (int rr = 0; rr < 4; ++rr) {
      const int e = ty + rr * 8;
      HbT[((size_t)b * DD + (size_t)(e0 + e)) * DD + (size_t)(d0 + tx)] = tile[tx][e];
    }
  }
}

// ===== Kernel 5: out[b] = V[b] @ H[b] (plain bf16 WMMA; H in [0,1]) =======
__global__ __launch_bounds__(256) void out_gemm(
    const unsigned short* __restrict__ Vh, const unsigned short* __restrict__ HbT,
    float* __restrict__ out) {
  const int lane = threadIdx.x & 31;
  const int wave = threadIdx.x >> 5;
  const int b = blockIdx.z;
  const int row0 = blockIdx.y * 128 + (wave >> 1) * 32;  // n within batch
  const int col0 = blockIdx.x * 64 + (wave & 1) * 32;    // e
  const size_t arow0 = (size_t)b * NN + (size_t)row0;
  const size_t hb = (size_t)b * DD * DD;

  AccF acc[2][2];
#pragma unroll
  for (int i = 0; i < 2; ++i)
#pragma unroll
    for (int j = 0; j < 2; ++j)
#pragma unroll
      for (int r = 0; r < 8; ++r) acc[i][j].f[r] = 0.0f;

  for (int k0 = 0; k0 < DD; k0 += 32) {  // contraction over d
    __builtin_prefetch(Vh + (arow0 + (size_t)(lane & 15)) * DD + (size_t)(k0 + 32), 0, 1);
    FragBF a[2], bf[2];
#pragma unroll
    for (int mt = 0; mt < 2; ++mt)
      loadA_rm1(a[mt], Vh, arow0 + (size_t)(mt * 16), k0, DD, lane);
#pragma unroll
    for (int nt = 0; nt < 2; ++nt)
      loadB_tr1(bf[nt], HbT, hb, k0, col0 + nt * 16, DD, lane);
#pragma unroll
    for (int mt = 0; mt < 2; ++mt)
#pragma unroll
      for (int nt = 0; nt < 2; ++nt)
        acc[mt][nt].v = wmma_bf16(a[mt], bf[nt], acc[mt][nt].v);
  }

#pragma unroll
  for (int mt = 0; mt < 2; ++mt)
#pragma unroll
    for (int nt = 0; nt < 2; ++nt)
      storeC(out, acc[mt][nt], arow0 + (size_t)(mt * 16), col0 + nt * 16, DD, lane, 1.0f);
}

// ===== Host-side orchestration ===========================================
extern "C" void kernel_launch(void* const* d_in, const int* in_sizes, int n_in,
                              void* d_out, int out_size, void* d_ws, size_t ws_size,
                              hipStream_t stream) {
  (void)in_sizes; (void)n_in; (void)out_size; (void)ws_size;
  const float* text = (const float*)d_in[0];
  const float* WQ   = (const float*)d_in[1];
  const float* Wk   = (const float*)d_in[2];
  const float* Wv   = (const float*)d_in[3];
  float* out = (float*)d_out;

  const size_t XSZ = (size_t)MROWS * DD;  // 16,777,216 elements
  const size_t WSZ = (size_t)DD * NC;     //  3,145,728 elements
  unsigned short* Xh  = (unsigned short*)d_ws;
  unsigned short* Xl  = Xh + XSZ;
  unsigned short* WhT = Xl + XSZ;
  unsigned short* WlT = WhT + WSZ;
  unsigned short* QhT = WlT + WSZ;
  unsigned short* QlT = QhT + XSZ;
  unsigned short* KhT = QlT + XSZ;
  unsigned short* KlT = KhT + XSZ;
  unsigned short* Vh  = KlT + XSZ;
  float*          Hf  = (float*)(void*)Xh;  // alias (X dead after GEMM1)
  unsigned short* HbT = Xl;                 // alias

  const int T = 256;
  cvt_split<<<(int)((XSZ + T - 1) / T), T, 0, stream>>>(text, Xh, Xl, (int)XSZ);
  cvt_w<<<(int)((WSZ + T - 1) / T), T, 0, stream>>>(WQ, Wk, Wv, WhT, WlT);

  qkv_gemm<<<dim3(NC / 64, MROWS / 128), T, 0, stream>>>(
      Xh, Xl, WhT, WlT, QhT, QlT, KhT, KlT, Vh);

  h_gemm<<<dim3(DD / 64, DD / 128, BB), T, 0, stream>>>(KhT, KlT, QhT, QlT, Hf);

  softmax_batch<<<(DD / 32) * (DD / 32), T, 0, stream>>>(Hf, HbT);

  out_gemm<<<dim3(DD / 64, NN / 128, BB), T, 0, stream>>>(Vh, HbT, out);
}
